// LSTMCell_57990648430624
// MI455X (gfx1250) — matrix operand order repaired and checked by reading, not compiled, not run on hardware
//
#include <hip/hip_runtime.h>
#include <hip/hip_bf16.h>

// ---- problem constants -----------------------------------------------------
#define BATCH 8192
#define DIN   2048
#define HID   2048
#define KTOT  4096   // D + H

// ---- tiling ----------------------------------------------------------------
#define BM 128       // rows of x/h0 per block
#define BN 64        // cols per gate per block (block covers 4*BN gate cols)
#define BK 64        // k per LDS stage (two wmma k-steps of 32)
#define LDA 72       // padded LDS row stride (ushorts): 144B = 16B-aligned, conflict-reducing
#define LDB 72

typedef __attribute__((ext_vector_type(16))) __bf16 v16bf;
typedef __attribute__((ext_vector_type(8)))  float  v8f;

struct Bits256 { uint4 lo, hi; };

// fp32 -> bf16 round-to-nearest-even (bit trick)
__device__ __forceinline__ unsigned short f2bf(float f) {
    unsigned int u = __float_as_uint(f);
    u += 0x7FFFu + ((u >> 16) & 1u);
    return (unsigned short)(u >> 16);
}

__device__ __forceinline__ float sigmoidf_(float x) {
    return 1.0f / (1.0f + __expf(-x));
}

// Generic (shared) pointer -> 32-bit LDS byte offset
__device__ __forceinline__ unsigned lds_off(const void* p) {
    return (unsigned)(unsigned long long)(const __attribute__((address_space(3))) void*)p;
}

// CDNA5 async copy: 16B global -> LDS, tracked by ASYNCcnt (ISA 10.x / 15.18.3)
__device__ __forceinline__ void async_ld_b128(unsigned lds_byte_off, const void* gaddr) {
    asm volatile("global_load_async_to_lds_b128 %0, %1, off"
                 :: "v"(lds_byte_off), "v"(gaddr)
                 : "memory");
}

#define WAIT_ASYNCCNT(n) asm volatile("s_wait_asynccnt " #n ::: "memory")

// A fragment 16x32 bf16 (ISA 7.12.2): lane m = lane&15;
// lanes 0-15 hold K = kb+{0..7, 16..23}; lanes 16-31 hold K = kb+{8..15, 24..31}
__device__ __forceinline__ v16bf load_a_frag(const unsigned short* sA, int mBase, int kb, int lane) {
    const int m  = mBase + (lane & 15);
    const int kk = kb + ((lane & 16) >> 1);          // +8 for upper half-wave
    const unsigned short* p = sA + m * LDA + kk;
    Bits256 t;
    t.lo = *(const uint4*)(p);
    t.hi = *(const uint4*)(p + 16);
    return __builtin_bit_cast(v16bf, t);
}

// B fragment 32x16 bf16: lane n = lane&15; lanes 0-15 hold K=kb+{0..7, 16..23},
// lanes 16-31 hold K=kb+{8..15, 24..31} (row of W)
__device__ __forceinline__ v16bf load_b_frag(const unsigned short* sB, int nBase, int kb, int lane) {
    const int n  = nBase + (lane & 15);
    const int kk = kb + ((lane & 16) >> 1);          // +8 for upper half-wave
    const unsigned short* p = sB + n * LDB + kk;
    Bits256 t;
    t.lo = *(const uint4*)(p);
    t.hi = *(const uint4*)(p + 16);
    return __builtin_bit_cast(v16bf, t);
}

// ============================================================================
// Pass 1: pack fp32 [lo | hi] rows into one bf16 row of length KTOT.
// dst[r][0:DIN] = lo[r][:], dst[r][DIN:KTOT] = hi[r][:]
// ============================================================================
__global__ void __launch_bounds__(256)
pack_bf16_cat(const float* __restrict__ lo, const float* __restrict__ hi,
              unsigned short* __restrict__ dst, int rows) {
    const size_t idx4   = (size_t)blockIdx.x * 256 + threadIdx.x;    // float4 units
    const size_t total4 = (size_t)rows * KTOT / 4;
    if (idx4 >= total4) return;
    const size_t e = idx4 * 4;
    const size_t r = e / KTOT;
    const int    k = (int)(e % KTOT);
    const float* src = (k < DIN) ? (lo + r * DIN + k) : (hi + r * HID + (k - DIN));
    const float4 v = *(const float4*)src;
    const unsigned int l = (unsigned)f2bf(v.x) | ((unsigned)f2bf(v.y) << 16);
    const unsigned int h = (unsigned)f2bf(v.z) | ((unsigned)f2bf(v.w) << 16);
    *(uint2*)(dst + e) = make_uint2(l, h);
}

// ============================================================================
// Pass 2: bf16 WMMA GEMM with async global->LDS staging + fused LSTM epilogue.
// gates[m, g*H + n] = sum_k A[m,k] * Wcat[g][n,k]
// ============================================================================
struct GemmParams {
    const unsigned short* __restrict__ Abf;   // [BATCH][KTOT] bf16
    const unsigned short* __restrict__ Wbf;   // [4][HID][KTOT] bf16
    const float* __restrict__ c0;
    const float* __restrict__ bx[4];
    const float* __restrict__ bh[4];
    float* __restrict__ h_out;
    float* __restrict__ c_out;
};

__global__ void __launch_bounds__(256)
lstm_gemm_async(GemmParams q) {
    __shared__ unsigned short sA[2][BM * LDA];        // 2 x 18 KB
    __shared__ unsigned short sB[2][4 * BN * LDB];    // 2 x 36 KB
    __shared__ float sBias[4][BN];

    const int tid  = threadIdx.x;
    const int lane = tid & 31;
    const int w    = tid >> 5;
    const int wm   = w >> 1;        // 0..3
    const int wn   = w & 1;         // 0..1
    const int rowBase = blockIdx.y * BM;
    const int colBase = blockIdx.x * BN;

    {
        const int g  = tid >> 6;
        const int jj = tid & 63;
        sBias[g][jj] = q.bx[g][colBase + jj] + q.bh[g][colBase + jj];
    }

    v8f acc[4][2][2];
    #pragma unroll
    for (int g = 0; g < 4; ++g)
        #pragma unroll
        for (int a = 0; a < 2; ++a)
            #pragma unroll
            for (int b = 0; b < 2; ++b)
                #pragma unroll
                for (int r = 0; r < 8; ++r)
                    acc[g][a][b][r] = 0.0f;

    // Issue one stage's async copies: 12 x b128 per thread (4 for A, 8 for B).
    auto issue_stage = [&](int s, int buf) {
        const int koff = s * BK;
        // A tile: 128 rows x 8 granules (16B each)
        #pragma unroll
        for (int i = 0; i < 4; ++i) {
            const int gi = i * 256 + tid;
            const int r  = gi >> 3;
            const int c  = gi & 7;
            async_ld_b128(lds_off(&sA[buf][r * LDA + c * 8]),
                          q.Abf + (size_t)(rowBase + r) * KTOT + koff + c * 8);
        }
        // B tiles: 4 gates x 64 rows x 8 granules
        #pragma unroll
        for (int i = 0; i < 8; ++i) {
            const int gi = i * 256 + tid;       // 0..2047
            const int g  = gi >> 9;
            const int rr = (gi >> 3) & 63;
            const int c  = gi & 7;
            async_ld_b128(lds_off(&sB[buf][(g * BN + rr) * LDB + c * 8]),
                          q.Wbf + ((size_t)g * HID + colBase + rr) * KTOT + koff + c * 8);
        }
    };

    const int NS = KTOT / BK;       // 64 stages
    issue_stage(0, 0);

    for (int s = 0; s < NS; ++s) {
        const int buf = s & 1;
        if (s + 1 < NS) {
            issue_stage(s + 1, buf ^ 1);
            WAIT_ASYNCCNT(12);      // stage s complete (async loads retire in order)
        } else {
            WAIT_ASYNCCNT(0);
        }
        __syncthreads();

        const unsigned short* cA = sA[buf];
        #pragma unroll
        for (int ks = 0; ks < 2; ++ks) {
            const int kb = ks * 32;
            const v16bf a0 = load_a_frag(cA, wm * 32 +  0, kb, lane);
            const v16bf a1 = load_a_frag(cA, wm * 32 + 16, kb, lane);
            #pragma unroll
            for (int g = 0; g < 4; ++g) {
                const unsigned short* cB = &sB[buf][g * BN * LDB];
                const v16bf b0 = load_b_frag(cB, wn * 32 +  0, kb, lane);
                const v16bf b1 = load_b_frag(cB, wn * 32 + 16, kb, lane);
                acc[g][0][0] = __builtin_amdgcn_wmma_f32_16x16x32_bf16(
                    false, a0, false, b0, (short)0, acc[g][0][0], false, false);
                acc[g][0][1] = __builtin_amdgcn_wmma_f32_16x16x32_bf16(
                    false, a0, false, b1, (short)0, acc[g][0][1], false, false);
                acc[g][1][0] = __builtin_amdgcn_wmma_f32_16x16x32_bf16(
                    false, a1, false, b0, (short)0, acc[g][1][0], false, false);
                acc[g][1][1] = __builtin_amdgcn_wmma_f32_16x16x32_bf16(
                    false, a1, false, b1, (short)0, acc[g][1][1], false, false);
            }
        }
        __syncthreads();
    }

    // Fused LSTM epilogue. C/D layout: VGPR r, lane l -> M=(l>=16)*8+r, N=l&15.
    #pragma unroll
    for (int mi = 0; mi < 2; ++mi) {
        #pragma unroll
        for (int ni = 0; ni < 2; ++ni) {
            const int jj = wn * 32 + ni * 16 + (lane & 15);
            const int j  = colBase + jj;
            const int rb = rowBase + wm * 32 + mi * 16 + ((lane >> 4) << 3);
            const float bi = sBias[0][jj];
            const float bf = sBias[1][jj];
            const float bg = sBias[2][jj];
            const float bo = sBias[3][jj];
            #pragma unroll
            for (int r = 0; r < 8; ++r) {
                const int row = rb + r;
                const float gi = acc[0][mi][ni][r] + bi;
                const float gf = acc[1][mi][ni][r] + bf;
                const float gg = acc[2][mi][ni][r] + bg;
                const float go = acc[3][mi][ni][r] + bo;
                const float it = sigmoidf_(gi);
                const float ft = sigmoidf_(gf);
                const float gt = tanhf(gg);
                const float ot = sigmoidf_(go);
                const size_t off = (size_t)row * HID + j;
                const float ct = ft * q.c0[off] + it * gt;
                const float ht = ot * tanhf(ct);
                q.h_out[off] = ht;
                q.c_out[off] = ct;
            }
        }
    }
}

// ============================================================================
// Fallback: round-1 fused kernel (fp32 loads + in-kernel bf16 conversion),
// used only if d_ws is too small for the bf16 copies.
// ============================================================================
struct Params {
    const float* __restrict__ x;
    const float* __restrict__ h0;
    const float* __restrict__ c0;
    const float* __restrict__ Wx[4];
    const float* __restrict__ Wh[4];
    const float* __restrict__ bx[4];
    const float* __restrict__ bh[4];
    float* __restrict__ h_out;
    float* __restrict__ c_out;
};

__global__ void __launch_bounds__(256)
lstm_cell_wmma_fused(Params p) {
    __shared__ unsigned short fA[BM * LDA];
    __shared__ unsigned short fB[4][BN * LDB];
    __shared__ float fBias[4][BN];

    const int tid  = threadIdx.x;
    const int lane = tid & 31;
    const int w    = tid >> 5;
    const int wm   = w >> 1;
    const int wn   = w & 1;
    const int rowBase = blockIdx.y * BM;
    const int colBase = blockIdx.x * BN;

    {
        const int g  = tid >> 6;
        const int jj = tid & 63;
        fBias[g][jj] = p.bx[g][colBase + jj] + p.bh[g][colBase + jj];
    }

    v8f acc[4][2][2];
    #pragma unroll
    for (int g = 0; g < 4; ++g)
        #pragma unroll
        for (int a = 0; a < 2; ++a)
            #pragma unroll
            for (int b = 0; b < 2; ++b)
                #pragma unroll
                for (int r = 0; r < 8; ++r)
                    acc[g][a][b][r] = 0.0f;

    for (int s = 0; s < KTOT / BK; ++s) {
        const int  ksrc      = s * BK;
        const bool firstHalf = (ksrc < DIN);
        const float* aSrc    = firstHalf ? p.x : p.h0;
        const int  koff      = firstHalf ? ksrc : (ksrc - DIN);

        #pragma unroll
        for (int it = 0; it < 8; ++it) {
            const int li = it * 256 + tid;
            const int r  = li >> 4;
            const int c4 = li & 15;
            const float4 v = *(const float4*)(aSrc + (size_t)(rowBase + r) * DIN + koff + c4 * 4);
            const unsigned int lo = (unsigned)f2bf(v.x) | ((unsigned)f2bf(v.y) << 16);
            const unsigned int hi = (unsigned)f2bf(v.z) | ((unsigned)f2bf(v.w) << 16);
            *(uint2*)&fA[r * LDA + c4 * 4] = make_uint2(lo, hi);
        }
        #pragma unroll
        for (int g = 0; g < 4; ++g) {
            const float* wSrc = firstHalf ? p.Wx[g] : p.Wh[g];
            #pragma unroll
            for (int it = 0; it < 4; ++it) {
                const int li = it * 256 + tid;
                const int r  = li >> 4;
                const int c4 = li & 15;
                const float4 v = *(const float4*)(wSrc + (size_t)(colBase + r) * DIN + koff + c4 * 4);
                const unsigned int lo = (unsigned)f2bf(v.x) | ((unsigned)f2bf(v.y) << 16);
                const unsigned int hi = (unsigned)f2bf(v.z) | ((unsigned)f2bf(v.w) << 16);
                *(uint2*)&fB[g][r * LDB + c4 * 4] = make_uint2(lo, hi);
            }
        }
        __syncthreads();

        #pragma unroll
        for (int ks = 0; ks < 2; ++ks) {
            const int kb = ks * 32;
            const v16bf a0 = load_a_frag(fA, wm * 32 +  0, kb, lane);
            const v16bf a1 = load_a_frag(fA, wm * 32 + 16, kb, lane);
            #pragma unroll
            for (int g = 0; g < 4; ++g) {
                const v16bf b0 = load_b_frag(fB[g], wn * 32 +  0, kb, lane);
                const v16bf b1 = load_b_frag(fB[g], wn * 32 + 16, kb, lane);
                acc[g][0][0] = __builtin_amdgcn_wmma_f32_16x16x32_bf16(
                    false, a0, false, b0, (short)0, acc[g][0][0], false, false);
                acc[g][0][1] = __builtin_amdgcn_wmma_f32_16x16x32_bf16(
                    false, a0, false, b1, (short)0, acc[g][0][1], false, false);
                acc[g][1][0] = __builtin_amdgcn_wmma_f32_16x16x32_bf16(
                    false, a1, false, b0, (short)0, acc[g][1][0], false, false);
                acc[g][1][1] = __builtin_amdgcn_wmma_f32_16x16x32_bf16(
                    false, a1, false, b1, (short)0, acc[g][1][1], false, false);
            }
        }
        __syncthreads();
    }

    #pragma unroll
    for (int mi = 0; mi < 2; ++mi) {
        #pragma unroll
        for (int ni = 0; ni < 2; ++ni) {
            const int jj = wn * 32 + ni * 16 + (lane & 15);
            const int j  = colBase + jj;
            const int rb = rowBase + wm * 32 + mi * 16 + ((lane >> 4) << 3);
            const float bi = fBias[0][jj];
            const float bf = fBias[1][jj];
            const float bg = fBias[2][jj];
            const float bo = fBias[3][jj];
            #pragma unroll
            for (int r = 0; r < 8; ++r) {
                const int row = rb + r;
                const float it = sigmoidf_(acc[0][mi][ni][r] + bi);
                const float ft = sigmoidf_(acc[1][mi][ni][r] + bf);
                const float gt = tanhf(acc[2][mi][ni][r] + bg);
                const float ot = sigmoidf_(acc[3][mi][ni][r] + bo);
                const size_t off = (size_t)row * HID + j;
                const float ct = ft * p.c0[off] + it * gt;
                const float ht = ot * tanhf(ct);
                p.h_out[off] = ht;
                p.c_out[off] = ct;
            }
        }
    }
}

// ============================================================================
extern "C" void kernel_launch(void* const* d_in, const int* in_sizes, int n_in,
                              void* d_out, int out_size, void* d_ws, size_t ws_size,
                              hipStream_t stream) {
    (void)in_sizes; (void)n_in; (void)out_size;

    const float* x  = (const float*)d_in[0];
    const float* h0 = (const float*)d_in[1];
    const float* c0 = (const float*)d_in[2];
    const float *Wx[4], *Wh[4], *bx[4], *bh[4];
    for (int g = 0; g < 4; ++g) {
        Wx[g] = (const float*)d_in[3  + 2 * g];
        bx[g] = (const float*)d_in[4  + 2 * g];
        Wh[g] = (const float*)d_in[11 + 2 * g];
        bh[g] = (const float*)d_in[12 + 2 * g];
    }
    float* out   = (float*)d_out;
    float* h_out = out;
    float* c_out = out + (size_t)BATCH * HID;

    const size_t needA = (size_t)BATCH * KTOT * sizeof(unsigned short);   // 64 MB
    const size_t needW = (size_t)4 * HID * KTOT * sizeof(unsigned short); // 64 MB

    dim3 grid(HID / BN, BATCH / BM);     // (32, 64)
    dim3 block(256);

    if (ws_size >= needA + needW) {
        unsigned short* Abf = (unsigned short*)d_ws;
        unsigned short* Wbf = Abf + (size_t)BATCH * KTOT;

        // Pass 1: pack A = [x | h0] and Wcat[g] = [Wx_g | Wh_g] to bf16.
        hipLaunchKernelGGL(pack_bf16_cat, dim3((unsigned)((size_t)BATCH * KTOT / 4 / 256)),
                           block, 0, stream, x, h0, Abf, BATCH);
        for (int g = 0; g < 4; ++g) {
            hipLaunchKernelGGL(pack_bf16_cat, dim3((unsigned)((size_t)HID * KTOT / 4 / 256)),
                               block, 0, stream, Wx[g], Wh[g],
                               Wbf + (size_t)g * HID * KTOT, HID);
        }

        // Pass 2: async-staged WMMA GEMM + fused epilogue.
        GemmParams q;
        q.Abf = Abf; q.Wbf = Wbf; q.c0 = c0;
        for (int g = 0; g < 4; ++g) { q.bx[g] = bx[g]; q.bh[g] = bh[g]; }
        q.h_out = h_out; q.c_out = c_out;
        hipLaunchKernelGGL(lstm_gemm_async, grid, block, 0, stream, q);
    } else {
        Params p;
        p.x = x; p.h0 = h0; p.c0 = c0;
        for (int g = 0; g < 4; ++g) { p.Wx[g] = Wx[g]; p.Wh[g] = Wh[g];
                                      p.bx[g] = bx[g]; p.bh[g] = bh[g]; }
        p.h_out = h_out; p.c_out = c_out;
        hipLaunchKernelGGL(lstm_cell_wmma_fused, grid, block, 0, stream, p);
    }
}
